// FeaturePropagation_22531398435386
// MI455X (gfx1250) — matrix operand order repaired and checked by reading, not compiled, not run on hardware
//
#include <hip/hip_runtime.h>
#include <hip/hip_bf16.h>

typedef __attribute__((ext_vector_type(16))) _Float16 v16h;
typedef __attribute__((ext_vector_type(8)))  _Float16 v8h;
typedef __attribute__((ext_vector_type(8)))  float    v8f;

// Problem constants
#define BB   16
#define N1   4096
#define N2   1024
#define C1   128
#define C2   256
#define CIN  384          // C2 + C1
#define NH   256          // hidden / output channels
#define MTOT (BB * N1)    // 65536 rows
#define BN_EPS 1e-3f

// ---------------------------------------------------------------------------
// Weight convert + transpose: W (K x N, f32 row-major) -> Wt (N x K, f16)
// ---------------------------------------------------------------------------
__global__ void convert_wt(const float* __restrict__ W, _Float16* __restrict__ Wt,
                           int K, int N) {
    int idx = blockIdx.x * blockDim.x + threadIdx.x;
    if (idx >= K * N) return;
    int k = idx / N;
    int n = idx - k * N;
    Wt[(size_t)n * K + k] = (_Float16)W[idx];
}

// ---------------------------------------------------------------------------
// 3-NN inverse-distance interpolation + concat -> X (MTOT x CIN, f16)
// grid: (B, N1/128), block: 128 threads (4 waves)
// ---------------------------------------------------------------------------
__global__ void interp_concat(const float* __restrict__ xyz1,
                              const float* __restrict__ xyz2,
                              const float* __restrict__ feat1,
                              const float* __restrict__ feat2,
                              _Float16* __restrict__ X) {
    __shared__ float sx[N2], sy[N2], sz[N2];
    __shared__ int   si[3][128];
    __shared__ float sw[3][128];

    const int b    = blockIdx.x;
    const int base = blockIdx.y * 128;     // point offset within batch
    const int tid  = threadIdx.x;

    // stage xyz2[b] into LDS
    for (int j = tid; j < N2; j += 128) {
        const float* p = xyz2 + ((size_t)b * N2 + j) * 3;
        sx[j] = p[0]; sy[j] = p[1]; sz[j] = p[2];
    }
    __syncthreads();

    // phase 1: per-thread top-3 nearest neighbors
    {
        const float* p = xyz1 + ((size_t)b * N1 + base + tid) * 3;
        const float px = p[0], py = p[1], pz = p[2];
        float d0 = 3.4e38f, d1 = 3.4e38f, d2 = 3.4e38f;
        int   i0 = 0, i1 = 0, i2 = 0;
        for (int j = 0; j < N2; ++j) {
            float dx = px - sx[j], dy = py - sy[j], dz = pz - sz[j];
            float d = dx * dx + dy * dy + dz * dz;
            if (d < d2) {
                if (d < d1) {
                    if (d < d0) { d2 = d1; i2 = i1; d1 = d0; i1 = i0; d0 = d; i0 = j; }
                    else        { d2 = d1; i2 = i1; d1 = d;  i1 = j; }
                } else          { d2 = d;  i2 = j; }
            }
        }
        float w0 = 1.0f / (d0 + 1e-10f);
        float w1 = 1.0f / (d1 + 1e-10f);
        float w2 = 1.0f / (d2 + 1e-10f);
        float inv = 1.0f / (w0 + w1 + w2);
        si[0][tid] = i0; si[1][tid] = i1; si[2][tid] = i2;
        sw[0][tid] = w0 * inv; sw[1][tid] = w1 * inv; sw[2][tid] = w2 * inv;
    }
    __syncthreads();

    // phase 2: wave-cooperative weighted gather of feat2 rows (coalesced)
    {
        const int wv   = tid >> 5;
        const int lane = tid & 31;
        for (int p = wv * 32; p < wv * 32 + 32; ++p) {
            const int j0 = si[0][p], j1 = si[1][p], j2 = si[2][p];
            const float w0 = sw[0][p], w1 = sw[1][p], w2 = sw[2][p];
            const float* f0 = feat2 + ((size_t)b * N2 + j0) * C2;
            const float* f1 = feat2 + ((size_t)b * N2 + j1) * C2;
            const float* f2 = feat2 + ((size_t)b * N2 + j2) * C2;
            _Float16* xr = X + ((size_t)b * N1 + base + p) * CIN;
            for (int c = lane; c < C2; c += 32)
                xr[c] = (_Float16)(w0 * f0[c] + w1 * f1[c] + w2 * f2[c]);
        }
    }

    // phase 3: copy feat1 into tail columns (coalesced over c)
    for (int i = tid; i < 128 * C1; i += 128) {
        int r = i >> 7;          // point within tile
        int c = i & (C1 - 1);    // channel
        X[((size_t)b * N1 + base + r) * CIN + C2 + c] =
            (_Float16)feat1[((size_t)b * N1 + base + r) * C1 + c];
    }
}

// ---------------------------------------------------------------------------
// WMMA fragment loads (per-lane), per the CDNA5 ISA VGPR layout tables.
// A (16x32 f16): lane%16 = row M; lanes<16 hold K {0..7,16..23}, lanes>=16
// hold K {8..15,24..31} -> two contiguous 8-half (16B) chunks at +0 and +16.
// ---------------------------------------------------------------------------
__device__ inline v16h load_frag_a(const _Float16* p) {
    union { v16h v; v8h h[2]; } u;
    u.h[0] = *(const v8h*)(p);
    u.h[1] = *(const v8h*)(p + 16);
    return u.v;
}
// B (32x16 f16) from column-major Wt: lane%16 = col N; lanes<16 hold K {0..7,16..23},
// lanes>=16 hold K {8..15,24..31} -> two 8-half chunks at +0 and +16.
__device__ inline v16h load_frag_b(const _Float16* p) {
    union { v16h v; v8h h[2]; } u;
    u.h[0] = *(const v8h*)(p);
    u.h[1] = *(const v8h*)(p + 16);
    return u.v;
}

__device__ inline v8f wmma_f16(v16h a, v16h b, v8f c) {
    return __builtin_amdgcn_wmma_f32_16x16x32_f16(false, a, false, b,
                                                  (short)0, c, false, false);
}

// ---------------------------------------------------------------------------
// Fused GEMM + bias + relu + BN(inference), software-pipelined K loop.
// A: MTOT x K (f16 row-major), Wt: NH x K (f16, column-major weights)
// out = s * max(A@W + bias, 0) + t,  s = gamma*rsqrt(var+eps), t = beta-mean*s
// block = 256 threads (8 waves); wave tile = 16(M) x 64(N)
// grid.x = MTOT/128, grid.y = NH/64
// ---------------------------------------------------------------------------
template <int K, bool F16OUT>
__global__ void gemm_bn_relu(const _Float16* __restrict__ A,
                             const _Float16* __restrict__ Wt,
                             const float* __restrict__ Bias,
                             const float* __restrict__ Gamma,
                             const float* __restrict__ Beta,
                             const float* __restrict__ Mean,
                             const float* __restrict__ Var,
                             void* __restrict__ Out) {
    const int wave  = threadIdx.x >> 5;
    const int lane  = threadIdx.x & 31;
    const int tileM = (blockIdx.x * 8 + wave) * 16;
    const int tileN = blockIdx.y * 64;

    const int m      = lane & 15;
    const int hiHalf = lane >> 4;        // 0 or 1
    const int koffA  = hiHalf << 3;      // 0 or 8
    const int koffB  = hiHalf << 3;      // 0 or 8

    const _Float16* Arow = A + (size_t)(tileM + m) * K + koffA;
    const _Float16* Bc0  = Wt + (size_t)(tileN +  0 + m) * K + koffB;
    const _Float16* Bc1  = Wt + (size_t)(tileN + 16 + m) * K + koffB;
    const _Float16* Bc2  = Wt + (size_t)(tileN + 32 + m) * K + koffB;
    const _Float16* Bc3  = Wt + (size_t)(tileN + 48 + m) * K + koffB;

    v8f acc[4] = {v8f{}, v8f{}, v8f{}, v8f{}};

    // software pipeline: fragments for step k are loaded one iteration ahead
    v16h a  = load_frag_a(Arow);
    v16h b0 = load_frag_b(Bc0);
    v16h b1 = load_frag_b(Bc1);
    v16h b2 = load_frag_b(Bc2);
    v16h b3 = load_frag_b(Bc3);

#pragma unroll
    for (int k0 = 32; k0 < K; k0 += 32) {
        v16h an  = load_frag_a(Arow + k0);
        v16h bn0 = load_frag_b(Bc0 + k0);
        v16h bn1 = load_frag_b(Bc1 + k0);
        v16h bn2 = load_frag_b(Bc2 + k0);
        v16h bn3 = load_frag_b(Bc3 + k0);
        acc[0] = wmma_f16(a, b0, acc[0]);
        acc[1] = wmma_f16(a, b1, acc[1]);
        acc[2] = wmma_f16(a, b2, acc[2]);
        acc[3] = wmma_f16(a, b3, acc[3]);
        a = an; b0 = bn0; b1 = bn1; b2 = bn2; b3 = bn3;
    }
    acc[0] = wmma_f16(a, b0, acc[0]);
    acc[1] = wmma_f16(a, b1, acc[1]);
    acc[2] = wmma_f16(a, b2, acc[2]);
    acc[3] = wmma_f16(a, b3, acc[3]);

    // epilogue: C/D layout -> VGPR r: row tileM + r + 8*hiHalf, col lane%16
    const int rbase = tileM + (hiHalf << 3);
#pragma unroll
    for (int t = 0; t < 4; ++t) {
        const int n  = tileN + t * 16 + m;
        const float bias = Bias[n];
        const float s    = Gamma[n] * rsqrtf(Var[n] + BN_EPS);
        const float tt   = Beta[n] - Mean[n] * s;
#pragma unroll
        for (int r = 0; r < 8; ++r) {
            float z = acc[t][r] + bias;
            z = z > 0.0f ? z : 0.0f;
            const float y = s * z + tt;
            const size_t off = (size_t)(rbase + r) * NH + n;
            if (F16OUT) ((_Float16*)Out)[off] = (_Float16)y;
            else        ((float*)Out)[off]    = y;
        }
    }
}

// ---------------------------------------------------------------------------
// launch
// ---------------------------------------------------------------------------
extern "C" void kernel_launch(void* const* d_in, const int* in_sizes, int n_in,
                              void* d_out, int out_size, void* d_ws, size_t ws_size,
                              hipStream_t stream) {
    const float* xyz1  = (const float*)d_in[0];
    const float* xyz2  = (const float*)d_in[1];
    const float* feat1 = (const float*)d_in[2];
    const float* feat2 = (const float*)d_in[3];
    const float* W0 = (const float*)d_in[4];
    const float* b0 = (const float*)d_in[5];
    const float* g0 = (const float*)d_in[6];
    const float* be0 = (const float*)d_in[7];
    const float* m0 = (const float*)d_in[8];
    const float* v0 = (const float*)d_in[9];
    const float* W1 = (const float*)d_in[10];
    const float* b1 = (const float*)d_in[11];
    const float* g1 = (const float*)d_in[12];
    const float* be1 = (const float*)d_in[13];
    const float* m1 = (const float*)d_in[14];
    const float* v1 = (const float*)d_in[15];

    char* ws = (char*)d_ws;
    constexpr size_t X_BYTES  = (size_t)MTOT * CIN * sizeof(_Float16); // 50,331,648
    constexpr size_t H_BYTES  = (size_t)MTOT * NH  * sizeof(_Float16); // 33,554,432
    constexpr size_t W0_BYTES = (size_t)CIN * NH * sizeof(_Float16);   //    196,608
    _Float16* X    = (_Float16*)(ws);
    _Float16* Hbuf = (_Float16*)(ws + X_BYTES);
    _Float16* W0t  = (_Float16*)(ws + X_BYTES + H_BYTES);
    _Float16* W1t  = (_Float16*)(ws + X_BYTES + H_BYTES + W0_BYTES);

    // weight convert + transpose
    convert_wt<<<(CIN * NH + 255) / 256, 256, 0, stream>>>(W0, W0t, CIN, NH);
    convert_wt<<<(NH * NH + 255) / 256, 256, 0, stream>>>(W1, W1t, NH, NH);

    // interpolation + concat -> X
    interp_concat<<<dim3(BB, N1 / 128), 128, 0, stream>>>(xyz1, xyz2, feat1, feat2, X);

    // layer 0: X(65536x384) @ W0 -> H f16
    gemm_bn_relu<CIN, true><<<dim3(MTOT / 128, NH / 64), 256, 0, stream>>>(
        X, W0t, b0, g0, be0, m0, v0, (void*)Hbuf);

    // layer 1: H(65536x256) @ W1 -> out f32
    gemm_bn_relu<NH, false><<<dim3(MTOT / 128, NH / 64), 256, 0, stream>>>(
        Hbuf, W1t, b1, g1, be1, m1, v1, d_out);
}
